// Renderer_72610717106497
// MI455X (gfx1250) — matrix operand order repaired and checked by reading, not compiled, not run on hardware
//
#include <hip/hip_runtime.h>

// Gaussian splat renderer for gfx1250 (MI455X, wave32, WMMA, async-LDS).
// Pipeline: preprocess -> bitonic depth sort -> gather (AoS geo + WMMA-B
// channel matrix) -> tiled WMMA compositing.

#define NG    1024
#define WRES  128
#define HRES  128
#define NPIX  (WRES * HRES)
#define USE_ASYNC_CHUNK_LOAD 1

typedef __attribute__((ext_vector_type(16))) _Float16 v16h;
typedef __attribute__((ext_vector_type(8)))  _Float16 v8h;
typedef __attribute__((ext_vector_type(8)))  float    v8f;

struct __align__(32) Geo {
  float px, py, ia, ib, ic, op, pad0, pad1;
};

__device__ __forceinline__ float sigmoidf(float x) { return 1.0f / (1.0f + __expf(-x)); }

// ---------------------------------------------------------------- kernel 1 --
__global__ void splat_preprocess(const float* __restrict__ means,
                                 const float* __restrict__ log_scales,
                                 const float* __restrict__ quats,
                                 const float* __restrict__ opl,
                                 const float* __restrict__ colors,
                                 const float* __restrict__ cam,
                                 const float* __restrict__ look,
                                 const float* __restrict__ upv,
                                 float* __restrict__ gpx, float* __restrict__ gpy,
                                 float* __restrict__ gia, float* __restrict__ gib,
                                 float* __restrict__ gic, float* __restrict__ gop,
                                 float* __restrict__ gdep,
                                 float* __restrict__ gcr, float* __restrict__ gcg,
                                 float* __restrict__ gcb) {
  int i = blockIdx.x * blockDim.x + threadIdx.x;
  if (i >= NG) return;

  // Camera basis (redundant per thread; trivial scalar work).
  float cx = cam[0], cy = cam[1], cz = cam[2];
  float fx0 = look[0] - cx, fy0 = look[1] - cy, fz0 = look[2] - cz;
  float inl = 1.0f / (sqrtf(fx0 * fx0 + fy0 * fy0 + fz0 * fz0) + 1e-8f);
  fx0 *= inl; fy0 *= inl; fz0 *= inl;
  float ux = upv[0], uy = upv[1], uz = upv[2];
  float rx = fy0 * uz - fz0 * uy, ry = fz0 * ux - fx0 * uz, rz = fx0 * uy - fy0 * ux;
  inl = 1.0f / (sqrtf(rx * rx + ry * ry + rz * rz) + 1e-8f);
  rx *= inl; ry *= inl; rz *= inl;
  float u2x = ry * fz0 - rz * fy0, u2y = rz * fx0 - rx * fz0, u2z = rx * fy0 - ry * fx0;
  inl = 1.0f / (sqrtf(u2x * u2x + u2y * u2y + u2z * u2z) + 1e-8f);
  u2x *= inl; u2y *= inl; u2z *= inl;

  float mx = means[3 * i] - cx, my = means[3 * i + 1] - cy, mz = means[3 * i + 2] - cz;
  float tx = rx * mx + ry * my + rz * mz;
  float ty = u2x * mx + u2y * my + u2z * mz;
  float tz = -(fx0 * mx + fy0 * my + fz0 * mz);
  float depth = -tz;

  const float foc = 1.7320508075688772f;  // 1/tan(30deg), aspect == 1
  float cw = (fabsf(-tz) < 1e-6f) ? 1e-6f : -tz;
  float px = (foc * tx / cw + 1.0f) * 0.5f * (float)WRES;
  float py = (foc * ty / cw + 1.0f) * 0.5f * (float)HRES;

  // quat -> rotation, scaled, 3D covariance
  float qw = quats[4 * i], qx = quats[4 * i + 1], qy = quats[4 * i + 2], qz = quats[4 * i + 3];
  inl = 1.0f / (sqrtf(qw * qw + qx * qx + qy * qy + qz * qz) + 1e-8f);
  qw *= inl; qx *= inl; qy *= inl; qz *= inl;
  float R[3][3] = {
      {1.0f - 2.0f * (qy * qy + qz * qz), 2.0f * (qx * qy - qw * qz), 2.0f * (qx * qz + qw * qy)},
      {2.0f * (qx * qy + qw * qz), 1.0f - 2.0f * (qx * qx + qz * qz), 2.0f * (qy * qz - qw * qx)},
      {2.0f * (qx * qz - qw * qy), 2.0f * (qy * qz + qw * qx), 1.0f - 2.0f * (qx * qx + qy * qy)}};
  float s[3] = {__expf(log_scales[3 * i]), __expf(log_scales[3 * i + 1]), __expf(log_scales[3 * i + 2])};
  float M[3][3], C3[3][3];
#pragma unroll
  for (int r = 0; r < 3; ++r)
#pragma unroll
    for (int c = 0; c < 3; ++c) M[r][c] = R[r][c] * s[c];
#pragma unroll
  for (int a = 0; a < 3; ++a)
#pragma unroll
    for (int b = 0; b < 3; ++b)
      C3[a][b] = M[a][0] * M[b][0] + M[a][1] * M[b][1] + M[a][2] * M[b][2];

  float zs = (fabsf(tz) < 1e-6f) ? 1e-6f : tz;
  const float fpix = foc * 0.5f * (float)WRES;  // fx == fy == foc*64
  float J00 = fpix / zs, J02 = -fpix * tx / (zs * zs);
  float J11 = fpix / zs, J12 = -fpix * ty / (zs * zs);
  float V3[3][3] = {{rx, ry, rz}, {u2x, u2y, u2z}, {-fx0, -fy0, -fz0}};
  float T2[2][3];
#pragma unroll
  for (int k = 0; k < 3; ++k) {
    T2[0][k] = J00 * V3[0][k] + J02 * V3[2][k];
    T2[1][k] = J11 * V3[1][k] + J12 * V3[2][k];
  }
  float TC[2][3];
#pragma unroll
  for (int r = 0; r < 2; ++r)
#pragma unroll
    for (int k = 0; k < 3; ++k)
      TC[r][k] = T2[r][0] * C3[0][k] + T2[r][1] * C3[1][k] + T2[r][2] * C3[2][k];
  float c00 = TC[0][0] * T2[0][0] + TC[0][1] * T2[0][1] + TC[0][2] * T2[0][2] + 0.3f;
  float c01 = TC[0][0] * T2[1][0] + TC[0][1] * T2[1][1] + TC[0][2] * T2[1][2];
  float c11 = TC[1][0] * T2[1][0] + TC[1][1] * T2[1][1] + TC[1][2] * T2[1][2] + 0.3f;
  float det = fmaxf(c00 * c11 - c01 * c01, 1e-6f);

  float oop = sigmoidf(opl[i]);
  if (!(depth > 0.1f && depth < 100.0f)) oop = 0.0f;  // identical to w=0 mask

  gpx[i] = px;  gpy[i] = py;
  gia[i] = c11 / det;  gib[i] = -c01 / det;  gic[i] = c00 / det;
  gop[i] = oop;  gdep[i] = depth;
  gcr[i] = sigmoidf(colors[3 * i]);
  gcg[i] = sigmoidf(colors[3 * i + 1]);
  gcb[i] = sigmoidf(colors[3 * i + 2]);
}

// ---------------------------------------------------------------- kernel 2 --
// Single-block bitonic sort of 1024 (depth, index) pairs, ascending depth.
__global__ void __launch_bounds__(NG) splat_sort(const float* __restrict__ dep,
                                                 int* __restrict__ order) {
  __shared__ float sk[NG];
  __shared__ int   sv[NG];
  int i = threadIdx.x;
  sk[i] = dep[i];
  sv[i] = i;
  __syncthreads();
  for (int k = 2; k <= NG; k <<= 1) {
    for (int j = k >> 1; j > 0; j >>= 1) {
      int ixj = i ^ j;
      if (ixj > i) {
        bool up = ((i & k) == 0);
        float a = sk[i], b = sk[ixj];
        if ((a > b) == up) {
          sk[i] = b; sk[ixj] = a;
          int t = sv[i]; sv[i] = sv[ixj]; sv[ixj] = t;
        }
      }
      __syncthreads();
    }
  }
  order[i] = sv[i];
}

// ---------------------------------------------------------------- kernel 3 --
// Gather into depth order: AoS geometry (32 B/gaussian) + per-chunk 32x16 f16
// channel matrix laid out exactly as the WMMA B fragment expects:
//   Bmat[chunk*512 + n*32 + kloc], lane reads a contiguous 32 B run.
__global__ void splat_gather(const int* __restrict__ order,
                             const float* __restrict__ gpx, const float* __restrict__ gpy,
                             const float* __restrict__ gia, const float* __restrict__ gib,
                             const float* __restrict__ gic, const float* __restrict__ gop,
                             const float* __restrict__ gdep,
                             const float* __restrict__ gcr, const float* __restrict__ gcg,
                             const float* __restrict__ gcb,
                             Geo* __restrict__ sgeo, _Float16* __restrict__ Bmat) {
  int t = blockIdx.x * blockDim.x + threadIdx.x;
  if (t >= NG) return;
  int o = order[t];
  Geo g;
  g.px = gpx[o];  g.py = gpy[o];
  g.ia = gia[o];  g.ib = gib[o];  g.ic = gic[o];
  g.op = gop[o];  g.pad0 = 0.0f;  g.pad1 = 0.0f;
  sgeo[t] = g;

  const int chunkbase = (t >> 5) << 9;  // chunk * 512 halves
  const int kloc = t & 31;
  _Float16 vals[4] = {(_Float16)gcr[o], (_Float16)gcg[o], (_Float16)gcb[o], (_Float16)gdep[o]};
#pragma unroll
  for (int n = 0; n < 16; ++n) {
    _Float16 bv = (n < 4) ? vals[n] : ((n == 4) ? (_Float16)1.0f : (_Float16)0.0f);
    Bmat[chunkbase + n * 32 + kloc] = bv;
  }
}

// ---------------------------------------------------------------- kernel 4 --
// 8 waves/block, 32 pixels/wave (one lane = one pixel, exact sequential
// front-to-back transmittance). Per 32-gaussian chunk:
//   geometry -> LDS via global_load_async_to_lds_b128 (ASYNCcnt),
//   contribs -> LDS staging as packed f16 (4x ds_store_b128),
//   A fragments: 2x ds_load_b128 each; B fragment: one 32 B global load;
//   2x v_wmma_f32_16x16x32_f16 accumulate [R,G,B,Depth,Alpha] per pixel.
__global__ void __launch_bounds__(256) splat_render(
    const Geo* __restrict__ sgeo, const _Float16* __restrict__ Bmat,
    float* __restrict__ out) {
  __shared__ Geo shgeo[32];
  __shared__ __align__(16) _Float16 stag[8][32][32];  // [wave][pixel][k], 16 KB

  const int tid  = threadIdx.x;
  const int lane = tid & 31;
  const int wv   = tid >> 5;
  const int pixBase = blockIdx.x * 256 + wv * 32;
  const int p = pixBase + lane;
  const float u = (float)(p & (WRES - 1)) + 0.5f;
  const float v = (float)(p >> 7) + 0.5f;

  float Tcur = 1.0f;               // running transmittance for this lane's pixel
  v8f acc0 = {};                   // pixels [0,16) of the wave, 16 channels
  v8f acc1 = {};                   // pixels [16,32)
  const int n = lane & 15;         // output channel / fragment row selector
  const int kbaseA = (lane < 16) ? 0 : 16;  // contiguous K-split for A (k = 16*(lane/16) + s)
  const int kbaseB = (lane < 16) ? 0 : 16;  // same contiguous K-split for B
  const int i0 = kbaseA >> 3;               // v8h index into a staging row

#if USE_ASYNC_CHUNK_LOAD
  const unsigned long long geobase = (unsigned long long)(uintptr_t)sgeo;
#endif

  for (int chunk = 0; chunk < NG; chunk += 32) {
    __syncthreads();
#if USE_ASYNC_CHUNK_LOAD
    if (tid < 32) {  // wave 0, full EXEC: 32 lanes x 32 B async copy to LDS
      unsigned int ldsa = (unsigned int)(uintptr_t)(&shgeo[tid]);   // LDS offset
      unsigned int voff = (unsigned int)((chunk + tid) * (int)sizeof(Geo));
      asm volatile(
          "global_load_async_to_lds_b128 %0, %1, %2\n\t"
          "global_load_async_to_lds_b128 %0, %1, %2 offset:16"
          :: "v"(ldsa), "v"(voff), "s"(geobase) : "memory");
      asm volatile("s_wait_asynccnt 0x0" ::: "memory");
    }
#else
    if (tid < 32) shgeo[tid] = sgeo[chunk + tid];
#endif
    if (lane == 0 && chunk + 32 < NG) {
      __builtin_prefetch(sgeo + chunk + 32, 0, 0);                 // global_prefetch_b8
      __builtin_prefetch(Bmat + (((chunk >> 5) + 1) << 9), 0, 0);
    }
    __syncthreads();

    // B fragment: contiguous, 32 B-aligned, coalesced across the wave.
    v16h bf = *(const v16h*)(Bmat + ((chunk >> 5) << 9) + (n << 5) + kbaseB);

    // Exact sequential compositing over the chunk; stage contribs as f16.
#pragma unroll
    for (int kk = 0; kk < 32; kk += 8) {
      v8h pk;
#pragma unroll
      for (int j = 0; j < 8; ++j) {
        const int k = kk + j;
        const float4 g0 = *(const float4*)&shgeo[k].px;  // px,py,ia,ib (broadcast)
        const float4 g1 = *(const float4*)&shgeo[k].ic;  // ic,op,--,--
        float dx = g0.x - u, dy = g0.y - v;
        float pw = -0.5f * (g0.z * dx * dx + 2.0f * g0.w * dx * dy + g1.x * dy * dy);
        float w = fminf(g1.y * __expf(fminf(pw, 0.0f)), 0.99f);
        pk[j] = (_Float16)(Tcur * w);
        Tcur *= (1.0f - w);
      }
      *(v8h*)&stag[wv][lane][kk] = pk;  // ds_store_b128
    }
    __threadfence_block();  // per-wave LDS staging visible before fragment reads

    // A fragments: lane's K-set is two contiguous 16 B runs per pixel row.
    const v8h* r0 = (const v8h*)&stag[wv][n][0];
    const v8h* r1 = (const v8h*)&stag[wv][16 + n][0];
    v8h a0l = r0[i0], a0h = r0[i0 + 1];
    v8h a1l = r1[i0], a1h = r1[i0 + 1];
    v16h a0 = __builtin_shufflevector(a0l, a0h, 0, 1, 2, 3, 4, 5, 6, 7, 8, 9, 10, 11, 12, 13, 14, 15);
    v16h a1 = __builtin_shufflevector(a1l, a1h, 0, 1, 2, 3, 4, 5, 6, 7, 8, 9, 10, 11, 12, 13, 14, 15);

    acc0 = __builtin_amdgcn_wmma_f32_16x16x32_f16(false, a0, false, bf, (short)0, acc0, false, false);
    acc1 = __builtin_amdgcn_wmma_f32_16x16x32_f16(false, a1, false, bf, (short)0, acc1, false, false);
  }

  // D layout: lane<16 -> M=j, lane>=16 -> M=j+8; channel = lane%16.
  if (n < 5) {
    const int rowoff = (lane >= 16) ? 8 : 0;
#pragma unroll
    for (int j = 0; j < 8; ++j) {
      int pl = j + rowoff;
      out[n * NPIX + pixBase + pl]      = acc0[j];
      out[n * NPIX + pixBase + 16 + pl] = acc1[j];
    }
  }
}

// ------------------------------------------------------------------ launch --
extern "C" void kernel_launch(void* const* d_in, const int* in_sizes, int n_in,
                              void* d_out, int out_size, void* d_ws, size_t ws_size,
                              hipStream_t stream) {
  (void)in_sizes; (void)n_in; (void)out_size; (void)ws_size;
  const float* means  = (const float*)d_in[0];
  const float* lsc    = (const float*)d_in[1];
  const float* quats  = (const float*)d_in[2];
  const float* opl    = (const float*)d_in[3];
  const float* colors = (const float*)d_in[4];
  const float* cam    = (const float*)d_in[5];
  const float* look   = (const float*)d_in[6];
  const float* up     = (const float*)d_in[7];
  float* out = (float*)d_out;

  float* ws = (float*)d_ws;
  float* gpx = ws + 0 * NG;   float* gpy = ws + 1 * NG;
  float* gia = ws + 2 * NG;   float* gib = ws + 3 * NG;
  float* gic = ws + 4 * NG;   float* gop = ws + 5 * NG;
  float* gdep = ws + 6 * NG;
  float* gcr = ws + 7 * NG;   float* gcg = ws + 8 * NG;  float* gcb = ws + 9 * NG;
  int*   order = (int*)(ws + 10 * NG);
  Geo*   sgeo  = (Geo*)(ws + 11 * NG);          // NG * 8 floats
  _Float16* Bmat = (_Float16*)(ws + 19 * NG);   // (NG/32) * 512 halves = 8K floats

  splat_preprocess<<<NG / 256, 256, 0, stream>>>(means, lsc, quats, opl, colors, cam, look, up,
                                                 gpx, gpy, gia, gib, gic, gop, gdep, gcr, gcg, gcb);
  splat_sort<<<1, NG, 0, stream>>>(gdep, order);
  splat_gather<<<NG / 256, 256, 0, stream>>>(order, gpx, gpy, gia, gib, gic, gop, gdep,
                                             gcr, gcg, gcb, sgeo, Bmat);
  splat_render<<<NPIX / 256, 256, 0, stream>>>(sgeo, Bmat, out);
}
